// NonLocalBlock_90194313216171
// MI455X (gfx1250) — matrix operand order repaired and checked by reading, not compiled, not run on hardware
//
#include <hip/hip_runtime.h>
#include <hip/hip_bf16.h>
#include <hip/hip_fp16.h>

typedef __attribute__((ext_vector_type(16))) _Float16 v16h;
typedef __attribute__((ext_vector_type(8)))  _Float16 v8h;
typedef __attribute__((ext_vector_type(8)))  float    v8f;

#define BATCH 4
#define CDIM  128
#define ICDIM 64      // head dim
#define NPOS  4096    // H*W
#define QBLK  128     // query rows per workgroup (8 waves x 16)
#define KBLK  64      // key rows staged per iteration
#define DHEAD 64

union Vu { v16h v; v8h h[2]; };

// Async 16B copy global -> LDS (ASYNCcnt-tracked, no VGPR round trip).
// CDNA5 ISA: GLOBAL_LOAD_ASYNC_TO_LDS_B128, GV mode:
// per-lane LDS byte offset in VDST vgpr, 64-bit global address in VADDR pair.
__device__ __forceinline__ void async_copy16(unsigned lds_off, const void* gptr) {
    asm volatile("global_load_async_to_lds_b128 %0, %1, off"
                 :: "v"(lds_off), "v"(gptr) : "memory");
}
__device__ __forceinline__ void wait_async0() {
    asm volatile("s_wait_asynccnt 0" ::: "memory");
}

// ---------------------------------------------------------------------------
// Kernel 1: 1x1-conv projections.  Q[b][n][o]=Wt.x+bt, K[b][n][o]=Wp.x+bp,
// V[b][o][n]=Wg.x+bg   (V kept transposed so the PV WMMA B-fragment is a
// contiguous LDS read).  0.8 GFLOP total -> plain VALU is fine.
// ---------------------------------------------------------------------------
__global__ __launch_bounds__(256) void proj_kernel(
    const float* __restrict__ x,
    const float* __restrict__ Wg, const float* __restrict__ bg,
    const float* __restrict__ Wt, const float* __restrict__ bt,
    const float* __restrict__ Wp, const float* __restrict__ bp,
    _Float16* __restrict__ Q, _Float16* __restrict__ K, _Float16* __restrict__ V)
{
    const size_t per_mat = (size_t)BATCH * NPOS * ICDIM;
    size_t idx = (size_t)blockIdx.x * blockDim.x + threadIdx.x;
    int mat = (int)(idx / per_mat);          // uniform per block (per_mat % 256 == 0)
    size_t r = idx % per_mat;

    const float* W; const float* bias;
    if (mat == 0)      { W = Wt; bias = bt; }   // theta -> Q
    else if (mat == 1) { W = Wp; bias = bp; }   // phi   -> K
    else               { W = Wg; bias = bg; }   // g     -> V

    int b, n, o;
    if (mat < 2) { // o fastest: W strided (cached), store coalesced
        o = (int)(r % ICDIM); size_t t = r / ICDIM; n = (int)(t % NPOS); b = (int)(t / NPOS);
    } else {       // n fastest: x + store coalesced, W broadcast
        n = (int)(r % NPOS);  size_t t = r / NPOS;  o = (int)(t % ICDIM); b = (int)(t / ICDIM);
    }

    const float* xp = x + (size_t)b * CDIM * NPOS + n;
    const float* wp = W + (size_t)o * CDIM;
    float acc = bias[o];
#pragma unroll 8
    for (int c = 0; c < CDIM; ++c) acc += wp[c] * xp[(size_t)c * NPOS];

    if (mat == 0)      Q[((size_t)b * NPOS + n) * ICDIM + o] = (_Float16)acc;
    else if (mat == 1) K[((size_t)b * NPOS + n) * ICDIM + o] = (_Float16)acc;
    else               V[((size_t)b * ICDIM + o) * NPOS + n] = (_Float16)acc;
}

// ---------------------------------------------------------------------------
// Kernel 2: flash-attention core.  Y[b][n][o] = softmax(Q K^T) V, f32 out.
// One wave = one 16-row Q tile; workgroup = 128 query rows.  K/V tiles are
// double-buffered in LDS and filled with async-to-LDS copies that overlap
// the 16 WMMAs + softmax of the current tile.
// ---------------------------------------------------------------------------
__global__ __launch_bounds__(256) void attn_kernel(
    const _Float16* __restrict__ Q,   // [B][N][64]
    const _Float16* __restrict__ K,   // [B][N][64]
    const _Float16* __restrict__ V,   // [B][64][N]  (= g, already transposed)
    float* __restrict__ Y)            // [B][N][64]
{
    __shared__ alignas(16) _Float16 Ks[2][KBLK * DHEAD];   // [key][dim]
    __shared__ alignas(16) _Float16 Vs[2][DHEAD * KBLK];   // [dim][key]
    __shared__ alignas(16) _Float16 Ps[8][16 * KBLK];      // per-wave P tile

    const int tid  = threadIdx.x;
    const int wave = tid >> 5;
    const int lane = tid & 31;
    const int hh   = lane >> 4;   // lane half (K/N sub-offset select)
    const int ln   = lane & 15;

    const int b    = blockIdx.x / (NPOS / QBLK);
    const int qblk = blockIdx.x % (NPOS / QBLK);
    const int q0   = qblk * QBLK + wave * 16;  // this wave's first query row

    // --- Q A-fragments, kept in registers for the whole kernel -------------
    // A fragment, this code's contiguous k order (16x32): lane ln holds row M=ln; halves i -> K=16h+i;
    // chunk c adds 32c.
    Vu qa[2];
    {
        const _Float16* qrow = Q + ((size_t)b * NPOS + q0 + ln) * DHEAD;
#pragma unroll
        for (int c = 0; c < 2; ++c) {
            qa[c].h[0] = *(const v8h*)(qrow + 32 * c + 16 * hh);
            qa[c].h[1] = *(const v8h*)(qrow + 32 * c + 16 * hh + 8);
        }
    }

    // Issue the async fills for one K/V tile (512 x 16B each, 2+2 per thread).
    auto stage = [&](int buf, int kb) {
        unsigned koff = (unsigned)(uintptr_t)(&Ks[buf][0]);
        unsigned voff = (unsigned)(uintptr_t)(&Vs[buf][0]);
#pragma unroll
        for (int i = 0; i < 2; ++i) {
            int j = tid + 256 * i;
            // K tile rows are contiguous in [B][N][64]
            const void* kg = (const void*)(K + ((size_t)b * NPOS + (size_t)kb * KBLK) * DHEAD + (size_t)j * 8);
            async_copy16(koff + (unsigned)j * 16u, kg);
            // V tile: 64 rows of 128B, row stride N halves
            int row = j >> 3, c16 = j & 7;
            const void* vg = (const void*)(V + ((size_t)b * DHEAD + row) * NPOS + (size_t)kb * KBLK + (size_t)c16 * 8);
            async_copy16(voff + (unsigned)j * 16u, vg);
        }
    };

    v8f O[4];
    const v8f vzero = {0.f, 0.f, 0.f, 0.f, 0.f, 0.f, 0.f, 0.f};
#pragma unroll
    for (int t = 0; t < 4; ++t) O[t] = vzero;
    float mrow[8], lrow[8];
#pragma unroll
    for (int r = 0; r < 8; ++r) { mrow[r] = -INFINITY; lrow[r] = 0.f; }

    const int NB = NPOS / KBLK;
    stage(0, 0);
    wait_async0();
    __syncthreads();

    int buf = 0;
    for (int kb = 0; kb < NB; ++kb) {
        // kick off next tile's fill; it overlaps this tile's compute
        if (kb + 1 < NB) stage(buf ^ 1, kb + 1);

        // ---- S = Q K^T : 4 column tiles x 2 K-chunks = 8 WMMAs ------------
        // B fragment in the same contiguous k order as A (32x16): lane ln holds col N=ln; halves i -> K=16h+i.
        v8f S[4];
#pragma unroll
        for (int t = 0; t < 4; ++t) {
            v8f acc = vzero;
#pragma unroll
            for (int c = 0; c < 2; ++c) {
                Vu bk;
                const _Float16* kp = Ks[buf] + (t * 16 + ln) * DHEAD + 32 * c + 16 * hh;
                bk.h[0] = *(const v8h*)(kp);
                bk.h[1] = *(const v8h*)(kp + 8);
                acc = __builtin_amdgcn_wmma_f32_16x16x32_f16(false, qa[c].v, false, bk.v,
                                                             (short)0, acc, false, false);
            }
            S[t] = acc;
        }

        // ---- online softmax.  C/D layout: VGPR r holds row (8*hh + r). ----
        float mnew[8];
#pragma unroll
        for (int r = 0; r < 8; ++r) {
            float mx = fmaxf(fmaxf(S[0][r], S[1][r]), fmaxf(S[2][r], S[3][r]));
#pragma unroll
            for (int s = 1; s < 16; s <<= 1) mx = fmaxf(mx, __shfl_xor(mx, s, 32));
            mnew[r] = fmaxf(mrow[r], mx);
        }
#pragma unroll
        for (int r = 0; r < 8; ++r) {
            float alpha = __expf(mrow[r] - mnew[r]);
            mrow[r] = mnew[r];
            lrow[r] *= alpha;
#pragma unroll
            for (int t = 0; t < 4; ++t) O[t][r] *= alpha;
            float psum = 0.f;
#pragma unroll
            for (int t = 0; t < 4; ++t) {
                float p = __expf(S[t][r] - mnew[r]);
                psum += p;
                Ps[wave][(8 * hh + r) * KBLK + t * 16 + ln] = (_Float16)p;
            }
#pragma unroll
            for (int s = 1; s < 16; s <<= 1) psum += __shfl_xor(psum, s, 32);
            lrow[r] += psum;
        }

        // ---- reload P as A-fragments (layout swap via wave-private LDS) ---
        Vu pa[2];
#pragma unroll
        for (int c = 0; c < 2; ++c) {
            const _Float16* pp = Ps[wave] + ln * KBLK + 32 * c + 16 * hh;
            pa[c].h[0] = *(const v8h*)(pp);
            pa[c].h[1] = *(const v8h*)(pp + 8);
        }

        // ---- O += P V : 4 dim tiles x 2 K-chunks = 8 WMMAs ----------------
#pragma unroll
        for (int t = 0; t < 4; ++t) {
#pragma unroll
            for (int c = 0; c < 2; ++c) {
                Vu bv;
                const _Float16* vp = Vs[buf] + (t * 16 + ln) * KBLK + 32 * c + 16 * hh;
                bv.h[0] = *(const v8h*)(vp);
                bv.h[1] = *(const v8h*)(vp + 8);
                O[t] = __builtin_amdgcn_wmma_f32_16x16x32_f16(false, pa[c].v, false, bv.v,
                                                              (short)0, O[t], false, false);
            }
        }

        // this wave's async fills done -> barrier makes them block-visible
        wait_async0();
        __syncthreads();
        buf ^= 1;
    }

    // ---- epilogue: divide by row sums, scatter to Y[b][n][o] --------------
#pragma unroll
    for (int r = 0; r < 8; ++r) {
        float inv = 1.0f / lrow[r];
        int row = q0 + 8 * hh + r;
        float* yp = Y + ((size_t)b * NPOS + row) * DHEAD;
#pragma unroll
        for (int t = 0; t < 4; ++t) yp[t * 16 + ln] = O[t][r] * inv;
    }
}

// ---------------------------------------------------------------------------
// Kernel 3: out[b][c][n] = x + bw[c] + sum_o Ww[c][o] * Y[b][n][o]
// ---------------------------------------------------------------------------
__global__ __launch_bounds__(256) void out_kernel(
    const float* __restrict__ x, const float* __restrict__ Ww,
    const float* __restrict__ bw, const float* __restrict__ Y,
    float* __restrict__ out)
{
    size_t idx = (size_t)blockIdx.x * blockDim.x + threadIdx.x;
    int n = (int)(idx % NPOS); size_t t = idx / NPOS;
    int c = (int)(t % CDIM);   int b = (int)(t / CDIM);
    const float* yp = Y + ((size_t)b * NPOS + n) * ICDIM;
    const float* wp = Ww + (size_t)c * ICDIM;
    float acc = bw[c];
#pragma unroll 8
    for (int o = 0; o < ICDIM; ++o) acc += wp[o] * yp[o];
    out[idx] = x[idx] + acc;
}

// ---------------------------------------------------------------------------
extern "C" void kernel_launch(void* const* d_in, const int* in_sizes, int n_in,
                              void* d_out, int out_size, void* d_ws, size_t ws_size,
                              hipStream_t stream) {
    (void)in_sizes; (void)n_in; (void)out_size; (void)ws_size;
    const float* x  = (const float*)d_in[0];
    const float* Wg = (const float*)d_in[1];
    const float* bg = (const float*)d_in[2];
    const float* Wt = (const float*)d_in[3];
    const float* bt = (const float*)d_in[4];
    const float* Wp = (const float*)d_in[5];
    const float* bp = (const float*)d_in[6];
    const float* Ww = (const float*)d_in[7];
    const float* bw = (const float*)d_in[8];
    float* out = (float*)d_out;

    const size_t per_mat = (size_t)BATCH * NPOS * ICDIM;   // 1,048,576
    _Float16* Qh = (_Float16*)d_ws;
    _Float16* Kh = Qh + per_mat;
    _Float16* Vh = Kh + per_mat;
    float*    Yf = (float*)(Vh + per_mat);                 // ws offset 6 MB, +4 MB

    dim3 blk(256);
    proj_kernel<<<dim3((unsigned)((3 * per_mat) / 256)), blk, 0, stream>>>(
        x, Wg, bg, Wt, bt, Wp, bp, Qh, Kh, Vh);
    attn_kernel<<<dim3((unsigned)(BATCH * NPOS / QBLK)), blk, 0, stream>>>(
        Qh, Kh, Vh, Yf);
    out_kernel<<<dim3((unsigned)(((size_t)BATCH * CDIM * NPOS) / 256)), blk, 0, stream>>>(
        x, Ww, bw, Yf, out);
}
